// TransformerLayerShard_76166950028537
// MI455X (gfx1250) — matrix operand order repaired and checked by reading, not compiled, not run on hardware
//
#include <hip/hip_runtime.h>
#include <stdint.h>

// ---------------------------------------------------------------------------
// Problem constants (from reference)
// ---------------------------------------------------------------------------
constexpr int BATCH = 2;
constexpr int SEQ   = 2048;
constexpr int DMODEL= 2048;
constexpr int NHEAD = 16;
constexpr int DPH   = 128;     // DMODEL / NHEAD
constexpr int DFF   = 8192;
constexpr int MROWS = BATCH * SEQ;     // 4096

#define DEVFN static __device__ __forceinline__
#define USE_ASYNC_COPY 1

// ---------------------------------------------------------------------------
// WMMA types (CDNA5 gfx1250, wave32)
// ---------------------------------------------------------------------------
typedef __attribute__((ext_vector_type(16))) __bf16 v16bf;
typedef __attribute__((ext_vector_type(8)))  float  v8f;

union FragA { v16bf v; uint4 q[2]; unsigned short us[16]; };
union FragB { v16bf v; uint4 q[2]; unsigned short us[16]; };

DEVFN v8f zero8() { v8f z = {0.f,0.f,0.f,0.f,0.f,0.f,0.f,0.f}; return z; }

DEVFN v8f wmma_bf16(const FragA& a, const FragB& b, v8f c) {
  // D = A(16x32 bf16) * B(32x16 bf16) + C(16x16 f32)
  return __builtin_amdgcn_wmma_f32_16x16x32_bf16(
      /*neg_a=*/false, a.v, /*neg_b=*/false, b.v,
      /*c_mod=*/(short)0, c, /*reuse_a=*/false, /*reuse_b=*/false);
}

struct FalseT { static constexpr bool value = false; };
struct TrueT  { static constexpr bool value = true;  };

// ---------------------------------------------------------------------------
// CDNA5 async global->LDS copy (ASYNCcnt-tracked DMA path)
// ---------------------------------------------------------------------------
#if USE_ASYNC_COPY
DEVFN void async_g2l_b128(unsigned int lds_off, const void* gaddr) {
  // per-lane: LDS[lds_off..+15] = MEM[gaddr..+15]
  asm volatile("global_load_async_to_lds_b128 %0, %1, off"
               :: "v"(lds_off), "v"(gaddr) : "memory");
}
DEVFN void wait_asynccnt0() {
#if defined(__has_builtin)
#if __has_builtin(__builtin_amdgcn_s_wait_asynccnt)
  __builtin_amdgcn_s_wait_asynccnt(0);
#else
  asm volatile("s_wait_asynccnt 0x0" ::: "memory");
#endif
#else
  asm volatile("s_wait_asynccnt 0x0" ::: "memory");
#endif
}
DEVFN unsigned int lds_off32(const void* p) {
  // generic pointer to __shared__: low 32 bits are the LDS byte offset
  return (unsigned int)(uintptr_t)p;
}
#endif

// ---------------------------------------------------------------------------
// bf16 <-> f32 helpers (RNE)
// ---------------------------------------------------------------------------
DEVFN unsigned short f2bf(float f) {
  union { float f; uint32_t u; } x; x.f = f;
  uint32_t r = 0x7FFFu + ((x.u >> 16) & 1u);
  return (unsigned short)((x.u + r) >> 16);
}
DEVFN float bf2f(unsigned short h) {
  union { uint32_t u; float f; } x; x.u = ((uint32_t)h) << 16;
  return x.f;
}

DEVFN float gelu_tanh(float x) {
  // tanh-approx GELU with hardware exp: tanh(u) = 1 - 2/(1+exp(2u))
  float u = 0.7978845608028654f * (x + 0.044715f * x * x * x);
  float t = 1.f - 2.f / (1.f + __expf(2.f * u));
  return 0.5f * x * (1.f + t);
}

// ---------------------------------------------------------------------------
// 1) Weight convert + transpose (tiled, coalesced): w[K][N] f32 -> wt[N][K] bf16
// ---------------------------------------------------------------------------
__global__ __launch_bounds__(256) void cvt_transpose_kernel(
    const float* __restrict__ w, unsigned short* __restrict__ wt, int K, int N) {
  __shared__ unsigned short tile[32][33];
  const int tx = threadIdx.x & 31;
  const int ty = threadIdx.x >> 5;           // 8 row-groups
  const int n0 = blockIdx.x * 32;
  const int k0 = blockIdx.y * 32;
#pragma unroll
  for (int r = 0; r < 32; r += 8)
    tile[r + ty][tx] = f2bf(w[(size_t)(k0 + r + ty) * N + n0 + tx]);
  __syncthreads();
#pragma unroll
  for (int r = 0; r < 32; r += 8)
    wt[(size_t)(n0 + r + ty) * K + k0 + tx] = tile[tx][r + ty];
}

// ---------------------------------------------------------------------------
// 2) LayerNorm: x[row][2048] f32 -> xn bf16
// ---------------------------------------------------------------------------
__global__ __launch_bounds__(256) void ln_kernel(
    const float* __restrict__ x, const float* __restrict__ g,
    const float* __restrict__ beta, unsigned short* __restrict__ xn) {
  int row = blockIdx.x;
  const float* xr = x + (size_t)row * DMODEL;
  float s = 0.f, s2 = 0.f;
  for (int i = threadIdx.x; i < DMODEL; i += 256) {
    float v = xr[i]; s += v; s2 += v * v;
  }
  for (int o = 16; o > 0; o >>= 1) {
    s  += __shfl_down(s,  o, 32);
    s2 += __shfl_down(s2, o, 32);
  }
  __shared__ float ps[8], ps2[8];
  int wid = threadIdx.x >> 5, ln = threadIdx.x & 31;
  if (ln == 0) { ps[wid] = s; ps2[wid] = s2; }
  __syncthreads();
  if (threadIdx.x == 0) {
    float a = 0.f, b = 0.f;
    for (int i = 0; i < 8; i++) { a += ps[i]; b += ps2[i]; }
    ps[0] = a; ps2[0] = b;
  }
  __syncthreads();
  float mean = ps[0] * (1.f / DMODEL);
  float var  = ps2[0] * (1.f / DMODEL) - mean * mean;
  float inv  = rsqrtf(var + 1e-5f);
  unsigned short* o = xn + (size_t)row * DMODEL;
  for (int i = threadIdx.x; i < DMODEL; i += 256)
    o[i] = f2bf((xr[i] - mean) * inv * g[i] + beta[i]);
}

// ---------------------------------------------------------------------------
// 3) Tiled bf16 WMMA GEMM, double-buffered LDS (1 barrier / K-step),
//    tiles staged with async global->LDS DMA (ASYNCcnt).
//    C[M,N] = A[M,K] * Bt[N,K]^T. Block tile 64x256, 8 waves (2x4),
//    wave tile 32x64, K step 32.
//    EPI 0: store bf16       EPI 1: bf16(gelu(acc+bias))
//    EPI 2: store f32        EPI 3: f32 out += acc + bias
// ---------------------------------------------------------------------------
template <int EPI>
__global__ __launch_bounds__(256) void gemm_kernel(
    const unsigned short* __restrict__ A,
    const unsigned short* __restrict__ Bt,
    unsigned short* __restrict__ Cb,
    float* __restrict__ Cf,
    const float* __restrict__ bias,
    int Mdim, int Ndim, int Kdim) {
  __shared__ unsigned short sA[2][64 * 32];
  __shared__ unsigned short sB[2][256 * 32];

  const int tid  = threadIdx.x;
  const int wid  = tid >> 5;
  const int lane = tid & 31;
  const int wm   = wid >> 2;       // 0..1
  const int wn   = wid & 3;        // 0..3
  const int ln16 = lane & 15;
  const int half = lane >> 4;
  const int blockM = blockIdx.y * 64;
  const int blockN = blockIdx.x * 256;

  // per-thread staging assignment
  const int ar = tid >> 2, ac = (tid & 3) * 8;             // A: 64 rows x 4 chunks
  const unsigned short* Aptr = &A[(size_t)(blockM + ar) * Kdim + ac];
  const unsigned short* Bptr = &Bt[(size_t)(blockN + tid) * Kdim];

  v8f acc[2][4];
#pragma unroll
  for (int mt = 0; mt < 2; mt++)
#pragma unroll
    for (int nt = 0; nt < 4; nt++) acc[mt][nt] = zero8();

  // ---- preload K-step 0 into buffer 0 ----
#if USE_ASYNC_COPY
  async_g2l_b128(lds_off32(&sA[0][ar * 32 + ac]), Aptr);
  async_g2l_b128(lds_off32(&sB[0][tid * 32]),     Bptr);
  async_g2l_b128(lds_off32(&sB[0][tid * 32 + 8]), Bptr + 8);
  wait_asynccnt0();
#else
  *(uint4*)&sA[0][ar * 32 + ac] = *(const uint4*)Aptr;
  *(uint4*)&sB[0][tid * 32]     = ((const uint4*)Bptr)[0];
  *(uint4*)&sB[0][tid * 32 + 8] = ((const uint4*)Bptr)[1];
#endif
  __syncthreads();

  int buf = 0;
  for (int kb = 0; kb < Kdim; kb += 32) {
    const bool has_next = (kb + 32) < Kdim;
    // ---- kick off DMA of next tile into the other buffer ----
    if (has_next) {
#if USE_ASYNC_COPY
      async_g2l_b128(lds_off32(&sA[buf ^ 1][ar * 32 + ac]), Aptr + kb + 32);
      async_g2l_b128(lds_off32(&sB[buf ^ 1][tid * 32]),     Bptr + kb + 32);
      async_g2l_b128(lds_off32(&sB[buf ^ 1][tid * 32 + 8]), Bptr + kb + 40);
#endif
    }

    // ---- fragments + 8 WMMAs from current buffer ----
    FragA af[2];
    FragB bfm[4];
#pragma unroll
    for (int mt = 0; mt < 2; mt++) {
      int r = wm * 32 + mt * 16 + ln16;
      af[mt].q[0] = *(const uint4*)&sA[buf][r * 32 + half * 8];
      af[mt].q[1] = *(const uint4*)&sA[buf][r * 32 + 16 + half * 8];
    }
#pragma unroll
    for (int nt = 0; nt < 4; nt++) {
      int n = wn * 64 + nt * 16 + ln16;
      bfm[nt].q[0] = *(const uint4*)&sB[buf][n * 32 + half * 8];
      bfm[nt].q[1] = *(const uint4*)&sB[buf][n * 32 + 16 + half * 8];
    }
#pragma unroll
    for (int mt = 0; mt < 2; mt++)
#pragma unroll
      for (int nt = 0; nt < 4; nt++)
        acc[mt][nt] = wmma_bf16(af[mt], bfm[nt], acc[mt][nt]);

    if (has_next) {
#if !USE_ASYNC_COPY
      uint4 a_reg  = *(const uint4*)(Aptr + kb + 32);
      uint4 b_reg0 = ((const uint4*)(Bptr + kb + 32))[0];
      uint4 b_reg1 = ((const uint4*)(Bptr + kb + 32))[1];
      *(uint4*)&sA[buf ^ 1][ar * 32 + ac] = a_reg;
      *(uint4*)&sB[buf ^ 1][tid * 32]     = b_reg0;
      *(uint4*)&sB[buf ^ 1][tid * 32 + 8] = b_reg1;
#else
      wait_asynccnt0();
#endif
      __syncthreads();
      buf ^= 1;
    }
  }

  // ---- epilogue ----
#pragma unroll
  for (int mt = 0; mt < 2; mt++)
#pragma unroll
    for (int nt = 0; nt < 4; nt++)
#pragma unroll
      for (int e = 0; e < 8; e++) {
        int m = blockM + wm * 32 + mt * 16 + e + 8 * half;  // C layout: M=e+8*half
        int n = blockN + wn * 64 + nt * 16 + ln16;          //           N=lane%16
        float v = acc[mt][nt][e];
        size_t idx = (size_t)m * Ndim + n;
        if (EPI == 0) {
          Cb[idx] = f2bf(v);
        } else if (EPI == 1) {
          Cb[idx] = f2bf(gelu_tanh(v + bias[n]));
        } else if (EPI == 2) {
          Cf[idx] = v;
        } else {
          Cf[idx] = Cf[idx] + v + bias[n];
        }
      }
}

// ---------------------------------------------------------------------------
// 4) RoPE in place on q and k (bf16, layout [row=b*S+s][h*128+d]).
//    q is additionally pre-scaled by 1/sqrt(DPH): RoPE is linear, q is only
//    consumed by Q*K^T, so this removes the scale from the attention loop.
// ---------------------------------------------------------------------------
__global__ __launch_bounds__(256) void rope_kernel(
    unsigned short* __restrict__ q, unsigned short* __restrict__ k) {
  size_t idx = (size_t)blockIdx.x * 256 + threadIdx.x;   // over 4096*1024 pairs
  if (idx >= (size_t)MROWS * (DMODEL / 2)) return;
  int row = (int)(idx >> 10);
  int pc  = (int)(idx & 1023);
  int s   = row & (SEQ - 1);
  int i   = pc & 63;   // pair index within head
  const float qscale = 0.08838834764831845f;  // 1/sqrt(128)
  float invf = expf(-0.14391156509f * (float)i);  // 10000^(-2i/128)
  float th = (float)s * invf;
  float sn = sinf(th), cs = cosf(th);
  size_t base = (size_t)row * DMODEL + 2 * (size_t)pc;
  float q0 = bf2f(q[base]), q1 = bf2f(q[base + 1]);
  q[base]     = f2bf((q0 * cs - q1 * sn) * qscale);
  q[base + 1] = f2bf((q1 * cs + q0 * sn) * qscale);
  float k0 = bf2f(k[base]), k1 = bf2f(k[base + 1]);
  k[base]     = f2bf(k0 * cs - k1 * sn);
  k[base + 1] = f2bf(k1 * cs + k0 * sn);
}

// ---------------------------------------------------------------------------
// 5) V transpose: v[b*S+s][h*128+d] -> vt[((b*H+h)*128+d)][s]
// ---------------------------------------------------------------------------
__global__ __launch_bounds__(256) void vtrans_kernel(
    const unsigned short* __restrict__ v, unsigned short* __restrict__ vt) {
  size_t idx = (size_t)blockIdx.x * 256 + threadIdx.x;
  if (idx >= (size_t)BATCH * NHEAD * DPH * SEQ) return;
  int s = (int)(idx & (SEQ - 1));
  size_t r = idx >> 11;
  int d = (int)(r & (DPH - 1)); r >>= 7;
  int h = (int)(r & (NHEAD - 1));
  int b = (int)(r >> 4);
  vt[idx] = v[((size_t)(b * SEQ + s)) * DMODEL + h * DPH + d];
}

// ---------------------------------------------------------------------------
// 6) Flash attention, one wave per (b, h, 32-query tile), causal + attn_bias.
//    Loop split: m0/32 unmasked key blocks + ONE peeled masked block
//    (mask offsets are loop-invariant there). Single pointer per stream,
//    all tile loads use 24-bit immediate offsets. Q is pre-scaled by 1/sqrt(d).
// ---------------------------------------------------------------------------
__global__ __launch_bounds__(32) void attn_kernel(
    const unsigned short* __restrict__ q,
    const unsigned short* __restrict__ k,
    const unsigned short* __restrict__ vt,
    const float* __restrict__ attn_bias,
    unsigned short* __restrict__ av) {
  __shared__ unsigned short sP[2][16 * 32];

  const int lane = threadIdx.x;
  const int ln16 = lane & 15;
  const int half = lane >> 4;
  int bid = blockIdx.x;
  const int mt = bid & 63;  bid >>= 6;   // SEQ/32 = 64 query tiles
  const int h  = bid & 15;  bid >>= 4;
  const int b  = bid;
  const int m0 = mt * 32;

  // Q fragments: 2 tiles x (16 rows x 128), 4 K-blocks of 32
  FragA qf[2][4];
#pragma unroll
  for (int qt = 0; qt < 2; qt++) {
    const unsigned short* qrow =
        q + ((size_t)(b * SEQ + m0 + qt * 16 + ln16)) * DMODEL + h * DPH;
#pragma unroll
    for (int kb = 0; kb < 4; kb++) {
      qf[qt][kb].q[0] = *(const uint4*)&qrow[kb * 32 + half * 8];
      qf[qt][kb].q[1] = *(const uint4*)&qrow[kb * 32 + 16 + half * 8];
    }
  }

  v8f O[2][8];
  float mi[2][8], li[2][8];
#pragma unroll
  for (int qt = 0; qt < 2; qt++) {
#pragma unroll
    for (int nt = 0; nt < 8; nt++) O[qt][nt] = zero8();
#pragma unroll
    for (int e = 0; e < 8; e++) { mi[qt][e] = -3.0e38f; li[qt][e] = 0.f; }
  }

  // single pointer per stream; per-block bump; loads use immediate offsets
  const unsigned short* kp =
      k + ((size_t)(b * SEQ + ln16)) * DMODEL + h * DPH + half * 8;
  const unsigned short* vp =
      vt + ((size_t)((b * NHEAD + h) * DPH + ln16)) * SEQ + half * 8;
  const float* bp = attn_bias + (size_t)(m0 + 8 * half) * SEQ + ln16;

  auto block_step = [&](auto MaskedTag) {
    constexpr bool MASKED = decltype(MaskedTag)::value;

    // ---- S = Q*K^T : 16 WMMAs; K fragments loaded per-kb (imm offsets) ----
    v8f st[2][2];
#pragma unroll
    for (int qt = 0; qt < 2; qt++) { st[qt][0] = zero8(); st[qt][1] = zero8(); }
#pragma unroll
    for (int kb = 0; kb < 4; kb++) {
      FragB kf0, kf1;
      kf0.q[0] = *(const uint4*)(kp + kb * 32);
      kf0.q[1] = *(const uint4*)(kp + kb * 32 + 16);
      kf1.q[0] = *(const uint4*)(kp + kb * 32 + 16 * DMODEL);
      kf1.q[1] = *(const uint4*)(kp + kb * 32 + 16 * DMODEL + 16);
#pragma unroll
      for (int qt = 0; qt < 2; qt++) {
        st[qt][0] = wmma_bf16(qf[qt][kb], kf0, st[qt][0]);
        st[qt][1] = wmma_bf16(qf[qt][kb], kf1, st[qt][1]);
      }
    }

    // ---- online softmax update ----
#pragma unroll
    for (int qt = 0; qt < 2; qt++)
#pragma unroll
      for (int e = 0; e < 8; e++) {
        float f0 = st[qt][0][e] + bp[(size_t)(qt * 16 + e) * SEQ];
        float f1 = st[qt][1][e] + bp[(size_t)(qt * 16 + e) * SEQ + 16];
        if (MASKED) {
          // loop-invariant causal offsets: n - m = ln16(+16) - (qt*16+e+8*half)
          int md = qt * 16 + e + 8 * half;
          f0 += (ln16      > md) ? -1e10f : 0.f;
          f1 += (ln16 + 16 > md) ? -1e10f : 0.f;
        }
        float mx = fmaxf(f0, f1);
#pragma unroll
        for (int o = 1; o < 16; o <<= 1) mx = fmaxf(mx, __shfl_xor(mx, o, 32));
        float mnew = fmaxf(mi[qt][e], mx);
        f0 = __expf(f0 - mnew);
        f1 = __expf(f1 - mnew);
        float c = __expf(mi[qt][e] - mnew);
        float rs = f0 + f1;
#pragma unroll
        for (int o = 1; o < 16; o <<= 1) rs += __shfl_xor(rs, o, 32);
        li[qt][e] = li[qt][e] * c + rs;
        mi[qt][e] = mnew;
        st[qt][0][e] = f0;           // st registers become P
        st[qt][1][e] = f1;
#pragma unroll
        for (int nt = 0; nt < 8; nt++) O[qt][nt][e] *= c;
      }

    // ---- transpose P (C layout -> A layout) through LDS ----
#pragma unroll
    for (int qt = 0; qt < 2; qt++)
#pragma unroll
      for (int e = 0; e < 8; e++) {
        sP[qt][(e + 8 * half) * 32 + ln16]      = f2bf(st[qt][0][e]);
        sP[qt][(e + 8 * half) * 32 + 16 + ln16] = f2bf(st[qt][1][e]);
      }
    __syncthreads();
    FragA pf[2];
#pragma unroll
    for (int qt = 0; qt < 2; qt++) {
      pf[qt].q[0] = *(const uint4*)&sP[qt][ln16 * 32 + half * 8];
      pf[qt].q[1] = *(const uint4*)&sP[qt][ln16 * 32 + 16 + half * 8];
    }
    __syncthreads();

    // ---- O += P * V : 16 WMMAs; V fragments reused by both query tiles ----
#pragma unroll
    for (int nt = 0; nt < 8; nt++) {
      FragB vf;
      vf.q[0] = *(const uint4*)(vp + nt * 16 * SEQ);
      vf.q[1] = *(const uint4*)(vp + nt * 16 * SEQ + 16);
#pragma unroll
      for (int qt = 0; qt < 2; qt++)
        O[qt][nt] = wmma_bf16(pf[qt], vf, O[qt][nt]);
    }

    kp += 32 * DMODEL;   // next 32 keys
    vp += 32;
    bp += 32;
  };

  const int nfull = m0 >> 5;             // fully-unmasked 32-key blocks
  for (int blk = 0; blk < nfull; blk++) block_step(FalseT{});
  block_step(TrueT{});                   // single masked block (keys m0..m0+31)

  // ---- normalize and store bf16 attn_vec ----
#pragma unroll
  for (int qt = 0; qt < 2; qt++) {
    unsigned short* outp =
        av + ((size_t)(b * SEQ + m0 + qt * 16)) * DMODEL + h * DPH;
#pragma unroll
    for (int nt = 0; nt < 8; nt++)
#pragma unroll
      for (int e = 0; e < 8; e++) {
        float o = O[qt][nt][e] / li[qt][e];
        int m = e + 8 * half;
        outp[(size_t)m * DMODEL + nt * 16 + ln16] = f2bf(o);
      }
  }
}

// ---------------------------------------------------------------------------
// Launcher
// ---------------------------------------------------------------------------
extern "C" void kernel_launch(void* const* d_in, const int* in_sizes, int n_in,
                              void* d_out, int out_size, void* d_ws,
                              size_t ws_size, hipStream_t stream) {
  (void)in_sizes; (void)n_in; (void)out_size; (void)ws_size;

  const float* x         = (const float*)d_in[0];
  const float* attn_bias = (const float*)d_in[1];
  const float* ln_scale  = (const float*)d_in[2];
  const float* ln_offset = (const float*)d_in[3];
  const float* wq        = (const float*)d_in[4];
  const float* wk        = (const float*)d_in[5];
  const float* wv        = (const float*)d_in[6];
  const float* wo        = (const float*)d_in[7];
  const float* w_fc_in   = (const float*)d_in[8];
  const float* b_fc_in   = (const float*)d_in[9];
  const float* w_fc_out  = (const float*)d_in[10];
  const float* b_fc_out  = (const float*)d_in[11];
  float* out = (float*)d_out;

  constexpr size_t SZ_ACT = (size_t)MROWS * DMODEL;
  constexpr size_t SZ_H   = (size_t)MROWS * DFF;
  constexpr size_t SZ_W   = (size_t)DMODEL * DMODEL;
  constexpr size_t SZ_WF  = (size_t)DMODEL * DFF;

  char* ws = (char*)d_ws;
  size_t off = 0;
  auto alloc = [&](size_t elems) {
    unsigned short* p = (unsigned short*)(ws + off);
    off += elems * 2;
    return p;
  };
  unsigned short* xn   = alloc(SZ_ACT);
  unsigned short* qb   = alloc(SZ_ACT);
  unsigned short* kb   = alloc(SZ_ACT);
  unsigned short* vb   = alloc(SZ_ACT);
  unsigned short* vtb  = alloc(SZ_ACT);
  unsigned short* avb  = alloc(SZ_ACT);
  unsigned short* hb   = alloc(SZ_H);
  unsigned short* wqt  = alloc(SZ_W);
  unsigned short* wkt  = alloc(SZ_W);
  unsigned short* wvt  = alloc(SZ_W);
  unsigned short* wot  = alloc(SZ_W);
  unsigned short* wfit = alloc(SZ_WF);
  unsigned short* wfot = alloc(SZ_WF);

  // 1) weight convert + transpose (tiled)
  {
    dim3 g(DMODEL / 32, DMODEL / 32);
    cvt_transpose_kernel<<<g, 256, 0, stream>>>(wq, wqt, DMODEL, DMODEL);
    cvt_transpose_kernel<<<g, 256, 0, stream>>>(wk, wkt, DMODEL, DMODEL);
    cvt_transpose_kernel<<<g, 256, 0, stream>>>(wv, wvt, DMODEL, DMODEL);
    cvt_transpose_kernel<<<g, 256, 0, stream>>>(wo, wot, DMODEL, DMODEL);
    dim3 gi(DFF / 32, DMODEL / 32);   // w_fc_in: K=DMODEL, N=DFF
    cvt_transpose_kernel<<<gi, 256, 0, stream>>>(w_fc_in, wfit, DMODEL, DFF);
    dim3 go(DMODEL / 32, DFF / 32);   // w_fc_out: K=DFF, N=DMODEL
    cvt_transpose_kernel<<<go, 256, 0, stream>>>(w_fc_out, wfot, DFF, DMODEL);
  }

  // 2) layernorm
  ln_kernel<<<MROWS, 256, 0, stream>>>(x, ln_scale, ln_offset, xn);

  // 3) QKV projections (bf16 out)
  {
    dim3 grid(DMODEL / 256, MROWS / 64);
    gemm_kernel<0><<<grid, 256, 0, stream>>>(xn, wqt, qb, nullptr, nullptr,
                                             MROWS, DMODEL, DMODEL);
    gemm_kernel<0><<<grid, 256, 0, stream>>>(xn, wkt, kb, nullptr, nullptr,
                                             MROWS, DMODEL, DMODEL);
    gemm_kernel<0><<<grid, 256, 0, stream>>>(xn, wvt, vb, nullptr, nullptr,
                                             MROWS, DMODEL, DMODEL);
  }

  // 4) RoPE on q,k (+ 1/sqrt(d) folded into q)
  {
    size_t pairs = (size_t)MROWS * (DMODEL / 2);
    rope_kernel<<<(int)((pairs + 255) / 256), 256, 0, stream>>>(qb, kb);
  }

  // 5) V transpose
  {
    size_t tot = (size_t)BATCH * NHEAD * DPH * SEQ;
    vtrans_kernel<<<(int)((tot + 255) / 256), 256, 0, stream>>>(vb, vtb);
  }

  // 6) flash attention (32 queries per wave)
  attn_kernel<<<BATCH * NHEAD * (SEQ / 32), 32, 0, stream>>>(qb, kb, vtb,
                                                             attn_bias, avb);

  // 7) attn_out = attn_vec @ wo -> d_out (f32 store)
  {
    dim3 grid(DMODEL / 256, MROWS / 64);
    gemm_kernel<2><<<grid, 256, 0, stream>>>(avb, wot, nullptr, out, nullptr,
                                             MROWS, DMODEL, DMODEL);
  }

  // 8) h = gelu(xn @ w_fc_in + b_fc_in)  (bf16)
  {
    dim3 grid(DFF / 256, MROWS / 64);
    gemm_kernel<1><<<grid, 256, 0, stream>>>(xn, wfit, hb, nullptr, b_fc_in,
                                             MROWS, DFF, DMODEL);
  }

  // 9) d_out += h @ w_fc_out + b_fc_out
  {
    dim3 grid(DMODEL / 256, MROWS / 64);
    gemm_kernel<3><<<grid, 256, 0, stream>>>(hb, wfot, nullptr, out, b_fc_out,
                                             MROWS, DMODEL, DFF);
  }
}
